// GraphBiasAttention_17875653886460
// MI455X (gfx1250) — matrix operand order repaired and checked by reading, not compiled, not run on hardware
//
#include <hip/hip_runtime.h>

// ---------------- problem constants ----------------
#define B_ 2
#define N_ 20000
#define C_ 512
#define S_ 128
#define H_ 8
#define D_ 64
#define E_ 320000
#define EPS_ 1e-6f

typedef __bf16 bf16;
typedef __bf16 v16bf __attribute__((ext_vector_type(16)));
typedef float  v8f   __attribute__((ext_vector_type(8)));

#define DEVI static __device__ __forceinline__

DEVI v8f zero8() {
  v8f z;
#pragma unroll
  for (int i = 0; i < 8; ++i) z[i] = 0.f;
  return z;
}

DEVI v8f wmma_bf16(v16bf a, v16bf b, v8f c) {
  return __builtin_amdgcn_wmma_f32_16x16x32_bf16(false, a, false, b, (short)0, c,
                                                 false, false);
}

// A fragment: 16x32 bf16, A row-major (M-major), lda elements (lda % 8 == 0).
// Lane layout (ISA 7.12.2): lanes 0-15 M=lane, k runs {half*8..+7, 16+half*8..+7}.
DEVI v16bf load_a_bf16(const bf16* __restrict__ A, int lda) {
  const int lane = threadIdx.x & 31;
  const int half = lane >> 4, m = lane & 15;
  const bf16* p = A + (size_t)m * lda + half * 8;
  union { uint4 u[2]; v16bf v; } U;
  U.u[0] = *reinterpret_cast<const uint4*>(p);
  U.u[1] = *reinterpret_cast<const uint4*>(p + 16);
  return U.v;
}

// Same fragment but sourced from f32 memory (converted on the fly).
DEVI v16bf load_a_f32(const float* __restrict__ A, int lda) {
  const int lane = threadIdx.x & 31;
  const int half = lane >> 4, m = lane & 15;
  const float* p = A + (size_t)m * lda + half * 8;
  v16bf v;
#pragma unroll
  for (int i = 0; i < 8; ++i) v[i] = (bf16)p[i];
#pragma unroll
  for (int i = 0; i < 8; ++i) v[8 + i] = (bf16)p[16 + i];
  return v;
}

// B fragment: 32x16 bf16 where B is stored TRANSPOSED (N-major), ldbt elements.
// Lane = column n, k runs {half*8..+7, 16+half*8..+7} (matches A fragment).
DEVI v16bf load_bt_bf16(const bf16* __restrict__ BT, int ldbt) {
  const int lane = threadIdx.x & 31;
  const int half = lane >> 4, n = lane & 15;
  const bf16* p = BT + (size_t)n * ldbt + half * 8;
  union { uint4 u[2]; v16bf v; } U;
  U.u[0] = *reinterpret_cast<const uint4*>(p);
  U.u[1] = *reinterpret_cast<const uint4*>(p + 16);
  return U.v;
}

// 32x32 output tile per wave: 4 accumulators, explicit 2-stage software
// pipeline so fragment loads for k+32 are in flight during the WMMAs for k.
DEVI void gemm_2x2(const bf16* __restrict__ A, const bf16* __restrict__ BT,
                   int K, int lda, int ldbt, v8f acc[4]) {
#pragma unroll
  for (int i = 0; i < 4; ++i) acc[i] = zero8();
  const bf16* A1 = A + (size_t)16 * lda;
  const bf16* B1 = BT + (size_t)16 * ldbt;
  v16bf a0 = load_a_bf16(A, lda),  a1 = load_a_bf16(A1, lda);
  v16bf b0 = load_bt_bf16(BT, ldbt), b1 = load_bt_bf16(B1, ldbt);
  for (int k = 32; k < K; k += 32) {
    __builtin_prefetch(A + k + 256, 0, 0);
    __builtin_prefetch(BT + k + 256, 0, 0);
    v16bf na0 = load_a_bf16(A + k, lda),  na1 = load_a_bf16(A1 + k, lda);
    v16bf nb0 = load_bt_bf16(BT + k, ldbt), nb1 = load_bt_bf16(B1 + k, ldbt);
    acc[0] = wmma_bf16(a0, b0, acc[0]);
    acc[1] = wmma_bf16(a0, b1, acc[1]);
    acc[2] = wmma_bf16(a1, b0, acc[2]);
    acc[3] = wmma_bf16(a1, b1, acc[3]);
    a0 = na0; a1 = na1; b0 = nb0; b1 = nb1;
  }
  acc[0] = wmma_bf16(a0, b0, acc[0]);
  acc[1] = wmma_bf16(a0, b1, acc[1]);
  acc[2] = wmma_bf16(a1, b0, acc[2]);
  acc[3] = wmma_bf16(a1, b1, acc[3]);
}

// ---------------- transpose + f32->bf16 convert ----------------
// in: (Z, R, Cc) f32 row-major; outT: (Z, Cc, R) bf16.
// grid: (Cc/32, R/32, Z), block (32, 8)
__global__ __launch_bounds__(256) void k_transpose_cvt(
    const float* __restrict__ in, bf16* __restrict__ outT, int R, int Cc) {
  __shared__ float tile[32][33];
  const size_t base = (size_t)blockIdx.z * R * Cc;
  const int c0 = blockIdx.x * 32, r0 = blockIdx.y * 32;
#pragma unroll
  for (int j = 0; j < 4; ++j) {
    int r = r0 + threadIdx.y + j * 8, c = c0 + threadIdx.x;
    if (r < R && c < Cc) tile[threadIdx.y + j * 8][threadIdx.x] = in[base + (size_t)r * Cc + c];
  }
  __syncthreads();
#pragma unroll
  for (int j = 0; j < 4; ++j) {
    int c = c0 + threadIdx.y + j * 8, r = r0 + threadIdx.x;
    if (r < R && c < Cc)
      outT[base + (size_t)c * R + r] = (bf16)tile[threadIdx.x][threadIdx.y + j * 8];
  }
}

// ---------------- stage 1: weights = softmax(x @ Ws + bs) ----------------
// One wave -> 16 rows x full S=128; all 8 B-fragments batched per k-step.
// grid (N/16, B), block 32.
__global__ __launch_bounds__(32) void k_weights(
    const float* __restrict__ x, const bf16* __restrict__ WsT,
    const float* __restrict__ bs, float* __restrict__ wf,
    bf16* __restrict__ wbf, bf16* __restrict__ wT, float* __restrict__ wsum) {
  const int b = blockIdx.y;
  const int row0 = blockIdx.x * 16;
  const int lane = threadIdx.x, half = lane >> 4, col = lane & 15;
  const float* Ab = x + ((size_t)b * N_ + row0) * C_;

  v8f acc[8];
#pragma unroll
  for (int t = 0; t < 8; ++t) acc[t] = zero8();
  for (int k = 0; k < C_; k += 32) {
    v16bf a = load_a_f32(Ab + k, C_);
    v16bf bfr[8];
#pragma unroll
    for (int t = 0; t < 8; ++t)
      bfr[t] = load_bt_bf16(WsT + (size_t)(t * 16) * C_ + k, C_);
#pragma unroll
    for (int t = 0; t < 8; ++t) acc[t] = wmma_bf16(a, bfr[t], acc[t]);
  }
#pragma unroll
  for (int t = 0; t < 8; ++t) {
    float bb = bs[t * 16 + col];
#pragma unroll
    for (int r = 0; r < 8; ++r) acc[t][r] += bb;
  }
  // row-wise softmax over 128 cols (cols live across 16 lanes x 8 accs)
#pragma unroll
  for (int r = 0; r < 8; ++r) {
    float mx = acc[0][r];
#pragma unroll
    for (int t = 1; t < 8; ++t) mx = fmaxf(mx, acc[t][r]);
#pragma unroll
    for (int off = 8; off; off >>= 1) mx = fmaxf(mx, __shfl_xor(mx, off, 32));
    float s = 0.f;
#pragma unroll
    for (int t = 0; t < 8; ++t) { float e = __expf(acc[t][r] - mx); acc[t][r] = e; s += e; }
#pragma unroll
    for (int off = 8; off; off >>= 1) s += __shfl_xor(s, off, 32);
    float inv = 1.f / s;
#pragma unroll
    for (int t = 0; t < 8; ++t) acc[t][r] *= inv;
  }
  // store f32, bf16 row-major, bf16 transposed; column sums via atomics
#pragma unroll
  for (int t = 0; t < 8; ++t) {
    const int scol = t * 16 + col;
    float part = 0.f;
#pragma unroll
    for (int r = 0; r < 8; ++r) {
      float v = acc[t][r];
      part += v;
      int row = row0 + half * 8 + r;
      size_t idx = ((size_t)b * N_ + row) * S_ + scol;
      wf[idx] = v;
      wbf[idx] = (bf16)v;
      wT[((size_t)b * S_ + scol) * N_ + row] = (bf16)v;
    }
    atomicAdd(&wsum[b * S_ + scol], part);
  }
}

// ---------------- stage 2: sparse aggregation ----------------
// one wave per edge; grid (E/8), block 256.
__global__ __launch_bounds__(256) void k_sparse(
    const int* __restrict__ rows, const int* __restrict__ cols,
    const float* __restrict__ vals, const float* __restrict__ wf,
    float* __restrict__ gw) {
  const int e = blockIdx.x * 8 + (threadIdx.x >> 5);
  const int lane = threadIdx.x & 31;
  if (e >= E_) return;
  const int r = rows[e], c = cols[e];
  const float v = vals[e];
#pragma unroll
  for (int b = 0; b < B_; ++b) {
    const float* wrow = wf + ((size_t)b * N_ + c) * S_;
    float* grow = gw + ((size_t)b * N_ + r) * S_;
#pragma unroll
    for (int j = 0; j < 4; ++j) {
      int s = lane + j * 32;
      atomicAdd(&grow[s], v * wrow[s]);
    }
  }
}

// ---------------- stage 3: slices = (wT @ x) / wsum  -> bf16 ----------------
// grid (S/32, C/32, B), block 32.
__global__ __launch_bounds__(32) void k_slices(
    const bf16* __restrict__ wT, const bf16* __restrict__ xT,
    const float* __restrict__ wsum, bf16* __restrict__ slc) {
  const int b = blockIdx.z, row0 = blockIdx.x * 32, col0 = blockIdx.y * 32;
  const bf16* A = wT + (size_t)b * S_ * N_ + (size_t)row0 * N_;
  const bf16* BT = xT + (size_t)b * C_ * N_ + (size_t)col0 * N_;
  v8f acc[4];
  gemm_2x2(A, BT, N_, N_, N_, acc);
  const int lane = threadIdx.x, half = lane >> 4, n = lane & 15;
#pragma unroll
  for (int i = 0; i < 2; ++i)
#pragma unroll
    for (int j = 0; j < 2; ++j)
#pragma unroll
      for (int r = 0; r < 8; ++r) {
        int row = row0 + i * 16 + half * 8 + r;
        float denom = fmaxf(wsum[b * S_ + row], EPS_);
        slc[((size_t)b * S_ + row) * C_ + col0 + j * 16 + n] =
            (bf16)(acc[i * 2 + j][r] / denom);
      }
}

// ---------------- stage 4: gb = wT @ gw ----------------
// grid (S/32, S/32, B), block 32.
__global__ __launch_bounds__(32) void k_gb(
    const bf16* __restrict__ wT, const bf16* __restrict__ gwT,
    float* __restrict__ gb) {
  const int b = blockIdx.z, row0 = blockIdx.x * 32, col0 = blockIdx.y * 32;
  const bf16* A = wT + (size_t)b * S_ * N_ + (size_t)row0 * N_;
  const bf16* BT = gwT + (size_t)b * S_ * N_ + (size_t)col0 * N_;
  v8f acc[4];
  gemm_2x2(A, BT, N_, N_, N_, acc);
  const int lane = threadIdx.x, half = lane >> 4, n = lane & 15;
#pragma unroll
  for (int i = 0; i < 2; ++i)
#pragma unroll
    for (int j = 0; j < 2; ++j)
#pragma unroll
      for (int r = 0; r < 8; ++r)
        gb[((size_t)b * S_ + row0 + i * 16 + half * 8 + r) * S_ + col0 + j * 16 + n] =
            acc[i * 2 + j][r];
}

// symmetrize + row-normalize + log; grid (B*S), block 128.
__global__ __launch_bounds__(128) void k_gb_norm(
    const float* __restrict__ gb, float* __restrict__ gbn) {
  const int b = blockIdx.x / S_, srow = blockIdx.x % S_, t = threadIdx.x;
  __shared__ float red[128];
  float v = 0.5f * (gb[((size_t)b * S_ + srow) * S_ + t] +
                    gb[((size_t)b * S_ + t) * S_ + srow]);
  red[t] = v;
  __syncthreads();
  for (int sft = 64; sft > 0; sft >>= 1) {
    if (t < sft) red[t] += red[t + sft];
    __syncthreads();
  }
  float denom = fmaxf(red[0], EPS_);
  gbn[((size_t)b * S_ + srow) * S_ + t] = logf(fmaxf(v / denom, EPS_));
}

// ---------------- stage 5: q/k/v head projections ----------------
// mode 0: q (scaled 1/sqrt(D), (B,H,S,D)); 1: k ((B,H,S,D)); 2: v -> vT (B,H,D,S).
// grid (S/32, C/32, B), block 32.
__global__ __launch_bounds__(32) void k_headproj(
    const bf16* __restrict__ slc, const bf16* __restrict__ WT,
    const float* __restrict__ bias, bf16* __restrict__ outp, int mode) {
  const int b = blockIdx.z, row0 = blockIdx.x * 32, col0 = blockIdx.y * 32;
  const bf16* A = slc + (size_t)b * S_ * C_ + (size_t)row0 * C_;
  const bf16* BT = WT + (size_t)col0 * C_;
  v8f acc[4];
  gemm_2x2(A, BT, C_, C_, C_, acc);
  const int lane = threadIdx.x, half = lane >> 4, n = lane & 15;
  const float scale = (mode == 0) ? 0.125f : 1.0f;  // 1/sqrt(64)
#pragma unroll
  for (int j = 0; j < 2; ++j) {
    const int colg = col0 + j * 16 + n, h = colg >> 6, d = colg & 63;
    const float bb = bias[colg];
#pragma unroll
    for (int i = 0; i < 2; ++i)
#pragma unroll
      for (int r = 0; r < 8; ++r) {
        int row = row0 + i * 16 + half * 8 + r;
        float v = (acc[i * 2 + j][r] + bb) * scale;
        if (mode == 2)
          outp[(((size_t)b * H_ + h) * D_ + d) * S_ + row] = (bf16)v;  // vT
        else
          outp[(((size_t)b * H_ + h) * S_ + row) * D_ + d] = (bf16)v;  // q / k
      }
  }
}

// ---------------- stage 6: attn = softmax(q kT + beta*gbn) ----------------
// grid (S/16, B*H), block 32. One wave -> 16 rows x 128 cols, batched B frags.
__global__ __launch_bounds__(32) void k_attn(
    const bf16* __restrict__ q, const bf16* __restrict__ kmat,
    const float* __restrict__ gbn, const float* __restrict__ beta_raw,
    bf16* __restrict__ attn) {
  const int bh = blockIdx.y, b = bh / H_;
  const int row0 = blockIdx.x * 16;
  const int lane = threadIdx.x, half = lane >> 4, col = lane & 15;
  const bf16* A = q + (size_t)bh * S_ * D_ + (size_t)row0 * D_;
  const bf16* KT = kmat + (size_t)bh * S_ * D_;
  const float beta = log1pf(__expf(beta_raw[0]));  // softplus
  v8f acc[8];
#pragma unroll
  for (int t = 0; t < 8; ++t) acc[t] = zero8();
#pragma unroll
  for (int kk = 0; kk < D_; kk += 32) {
    v16bf a = load_a_bf16(A + kk, D_);
    v16bf bfr[8];
#pragma unroll
    for (int t = 0; t < 8; ++t)
      bfr[t] = load_bt_bf16(KT + (size_t)(t * 16) * D_ + kk, D_);
#pragma unroll
    for (int t = 0; t < 8; ++t) acc[t] = wmma_bf16(a, bfr[t], acc[t]);
  }
#pragma unroll
  for (int t = 0; t < 8; ++t)
#pragma unroll
    for (int r = 0; r < 8; ++r) {
      int row = row0 + half * 8 + r;
      acc[t][r] += beta * gbn[((size_t)b * S_ + row) * S_ + t * 16 + col];
    }
#pragma unroll
  for (int r = 0; r < 8; ++r) {
    float mx = acc[0][r];
#pragma unroll
    for (int t = 1; t < 8; ++t) mx = fmaxf(mx, acc[t][r]);
#pragma unroll
    for (int off = 8; off; off >>= 1) mx = fmaxf(mx, __shfl_xor(mx, off, 32));
    float s = 0.f;
#pragma unroll
    for (int t = 0; t < 8; ++t) { float e = __expf(acc[t][r] - mx); acc[t][r] = e; s += e; }
#pragma unroll
    for (int off = 8; off; off >>= 1) s += __shfl_xor(s, off, 32);
    float inv = 1.f / s;
#pragma unroll
    for (int t = 0; t < 8; ++t) acc[t][r] *= inv;
  }
#pragma unroll
  for (int t = 0; t < 8; ++t)
#pragma unroll
    for (int r = 0; r < 8; ++r)
      attn[((size_t)bh * S_ + row0 + half * 8 + r) * S_ + t * 16 + col] =
          (bf16)acc[t][r];
}

// ---------------- stage 7: so = attn @ v (reassemble heads) ----------------
// grid (S/32, D/32, B*H), block 32.
__global__ __launch_bounds__(32) void k_av(
    const bf16* __restrict__ attn, const bf16* __restrict__ vT,
    bf16* __restrict__ sob) {
  const int bh = blockIdx.z, b = bh / H_, h = bh % H_;
  const int row0 = blockIdx.x * 32, col0 = blockIdx.y * 32;
  const bf16* A = attn + (size_t)bh * S_ * S_ + (size_t)row0 * S_;
  const bf16* BT = vT + (size_t)bh * D_ * S_ + (size_t)col0 * S_;
  v8f acc[4];
  gemm_2x2(A, BT, S_, S_, S_, acc);
  const int lane = threadIdx.x, half = lane >> 4, n = lane & 15;
#pragma unroll
  for (int i = 0; i < 2; ++i)
#pragma unroll
    for (int j = 0; j < 2; ++j)
#pragma unroll
      for (int r = 0; r < 8; ++r) {
        int row = row0 + i * 16 + half * 8 + r;
        sob[((size_t)b * S_ + row) * C_ + h * D_ + col0 + j * 16 + n] =
            (bf16)acc[i * 2 + j][r];
      }
}

// ---------------- stage 8: slices_out = so @ Wo + bo  (store transposed) ----
// grid (S/32, C/32, B), block 32.
__global__ __launch_bounds__(32) void k_wo(
    const bf16* __restrict__ sob, const bf16* __restrict__ WoT,
    const float* __restrict__ bo, bf16* __restrict__ so2T) {
  const int b = blockIdx.z, row0 = blockIdx.x * 32, col0 = blockIdx.y * 32;
  const bf16* A = sob + (size_t)b * S_ * C_ + (size_t)row0 * C_;
  const bf16* BT = WoT + (size_t)col0 * C_;
  v8f acc[4];
  gemm_2x2(A, BT, C_, C_, C_, acc);
  const int lane = threadIdx.x, half = lane >> 4, n = lane & 15;
#pragma unroll
  for (int j = 0; j < 2; ++j) {
    const int colg = col0 + j * 16 + n;
    const float bb = bo[colg];
#pragma unroll
    for (int i = 0; i < 2; ++i)
#pragma unroll
      for (int r = 0; r < 8; ++r) {
        int row = row0 + i * 16 + half * 8 + r;
        so2T[((size_t)b * C_ + colg) * S_ + row] = (bf16)(acc[i * 2 + j][r] + bb);
      }
  }
}

// ---------------- stage 9: out = weights @ slices_out ----------------
// grid (N/32, C/32, B), block 32.
__global__ __launch_bounds__(32) void k_final(
    const bf16* __restrict__ wbf, const bf16* __restrict__ so2T,
    float* __restrict__ outp) {
  const int b = blockIdx.z, row0 = blockIdx.x * 32, col0 = blockIdx.y * 32;
  const bf16* A = wbf + ((size_t)b * N_ + row0) * S_;
  const bf16* BT = so2T + (size_t)b * C_ * S_ + (size_t)col0 * S_;
  v8f acc[4];
  gemm_2x2(A, BT, S_, S_, S_, acc);
  const int lane = threadIdx.x, half = lane >> 4, n = lane & 15;
#pragma unroll
  for (int i = 0; i < 2; ++i)
#pragma unroll
    for (int j = 0; j < 2; ++j)
#pragma unroll
      for (int r = 0; r < 8; ++r)
        outp[((size_t)b * N_ + row0 + i * 16 + half * 8 + r) * C_ + col0 + j * 16 + n] =
            acc[i * 2 + j][r];
}

// ---------------- host side ----------------
extern "C" void kernel_launch(void* const* d_in, const int* in_sizes, int n_in,
                              void* d_out, int out_size, void* d_ws, size_t ws_size,
                              hipStream_t stream) {
  const float* x        = (const float*)d_in[0];
  const int*   adj      = (const int*)d_in[1];   // (2, E): rows then cols
  const float* adj_vals = (const float*)d_in[2];
  const float* Ws = (const float*)d_in[3];  const float* bs = (const float*)d_in[4];
  const float* Wq = (const float*)d_in[5];  const float* bq = (const float*)d_in[6];
  const float* Wk = (const float*)d_in[7];  const float* bk = (const float*)d_in[8];
  const float* Wv = (const float*)d_in[9];  const float* bv = (const float*)d_in[10];
  const float* Wo = (const float*)d_in[11]; const float* bo = (const float*)d_in[12];
  const float* beta_raw = (const float*)d_in[13];
  float* outp = (float*)d_out;

  char* wsb = (char*)d_ws;
  size_t off = 0;
  auto alloc = [&](size_t bytes) -> char* {
    char* p = wsb + off;
    off = (off + bytes + 255) & ~(size_t)255;
    return p;
  };
  bf16* xT   = (bf16*)alloc((size_t)B_ * N_ * C_ * 2);
  bf16* WsT  = (bf16*)alloc((size_t)C_ * S_ * 2);
  bf16* WqT  = (bf16*)alloc((size_t)C_ * C_ * 2);
  bf16* WkT  = (bf16*)alloc((size_t)C_ * C_ * 2);
  bf16* WvT  = (bf16*)alloc((size_t)C_ * C_ * 2);
  bf16* WoT  = (bf16*)alloc((size_t)C_ * C_ * 2);
  float* wf  = (float*)alloc((size_t)B_ * N_ * S_ * 4);
  bf16* wbf  = (bf16*)alloc((size_t)B_ * N_ * S_ * 2);
  bf16* wT   = (bf16*)alloc((size_t)B_ * N_ * S_ * 2);
  float* wsum = (float*)alloc((size_t)B_ * S_ * 4);
  float* gw  = (float*)alloc((size_t)B_ * N_ * S_ * 4);
  bf16* gwT  = (bf16*)alloc((size_t)B_ * N_ * S_ * 2);
  bf16* slc  = (bf16*)alloc((size_t)B_ * S_ * C_ * 2);
  float* gb  = (float*)alloc((size_t)B_ * S_ * S_ * 4);
  float* gbn = (float*)alloc((size_t)B_ * S_ * S_ * 4);
  bf16* qb   = (bf16*)alloc((size_t)B_ * H_ * S_ * D_ * 2);
  bf16* kb   = (bf16*)alloc((size_t)B_ * H_ * S_ * D_ * 2);
  bf16* vT   = (bf16*)alloc((size_t)B_ * H_ * D_ * S_ * 2);
  bf16* attn = (bf16*)alloc((size_t)B_ * H_ * S_ * S_ * 2);
  bf16* sob  = (bf16*)alloc((size_t)B_ * S_ * C_ * 2);
  bf16* so2T = (bf16*)alloc((size_t)B_ * C_ * S_ * 2);

  hipMemsetAsync(wsum, 0, (size_t)B_ * S_ * 4, stream);
  hipMemsetAsync(gw, 0, (size_t)B_ * N_ * S_ * 4, stream);

  const dim3 tb(32, 8);
  k_transpose_cvt<<<dim3(C_ / 32, N_ / 32, B_), tb, 0, stream>>>(x, xT, N_, C_);
  k_transpose_cvt<<<dim3(S_ / 32, C_ / 32, 1), tb, 0, stream>>>(Ws, WsT, C_, S_);
  k_transpose_cvt<<<dim3(C_ / 32, C_ / 32, 1), tb, 0, stream>>>(Wq, WqT, C_, C_);
  k_transpose_cvt<<<dim3(C_ / 32, C_ / 32, 1), tb, 0, stream>>>(Wk, WkT, C_, C_);
  k_transpose_cvt<<<dim3(C_ / 32, C_ / 32, 1), tb, 0, stream>>>(Wv, WvT, C_, C_);
  k_transpose_cvt<<<dim3(C_ / 32, C_ / 32, 1), tb, 0, stream>>>(Wo, WoT, C_, C_);

  k_weights<<<dim3(N_ / 16, B_), 32, 0, stream>>>(x, WsT, bs, wf, wbf, wT, wsum);
  k_sparse<<<E_ / 8, 256, 0, stream>>>(adj, adj + E_, adj_vals, wf, gw);
  k_transpose_cvt<<<dim3(S_ / 32, N_ / 32, B_), tb, 0, stream>>>(gw, gwT, N_, S_);

  k_slices<<<dim3(S_ / 32, C_ / 32, B_), 32, 0, stream>>>(wT, xT, wsum, slc);
  k_gb<<<dim3(S_ / 32, S_ / 32, B_), 32, 0, stream>>>(wT, gwT, gb);
  k_gb_norm<<<B_ * S_, 128, 0, stream>>>(gb, gbn);

  k_headproj<<<dim3(S_ / 32, C_ / 32, B_), 32, 0, stream>>>(slc, WqT, bq, qb, 0);
  k_headproj<<<dim3(S_ / 32, C_ / 32, B_), 32, 0, stream>>>(slc, WkT, bk, kb, 1);
  k_headproj<<<dim3(S_ / 32, C_ / 32, B_), 32, 0, stream>>>(slc, WvT, bv, vT, 2);

  k_attn<<<dim3(S_ / 16, B_ * H_), 32, 0, stream>>>(qb, kb, gbn, beta_raw, attn);
  k_av<<<dim3(S_ / 32, D_ / 32, B_ * H_), 32, 0, stream>>>(attn, vT, sob);
  k_wo<<<dim3(S_ / 32, C_ / 32, B_), 32, 0, stream>>>(sob, WoT, bo, so2T);
  k_final<<<dim3(N_ / 32, C_ / 32, B_), 32, 0, stream>>>(wbf, so2T, outp);

  (void)in_sizes; (void)n_in; (void)out_size; (void)ws_size;
}
